// BiEncode_51874615001232
// MI455X (gfx1250) — matrix operand order repaired and checked by reading, not compiled, not run on hardware
//
#include <hip/hip_runtime.h>
#include <hip/hip_bf16.h>
#include <stdint.h>

#define T_FRAMES 26
#define HIDDEN   512
#define INPUT    1024
#define BATCH    2048
#define GATES    2048           // 4*HIDDEN
#define KTOT     1536           // INPUT + HIDDEN (concatenated K)
#define MTILE    32             // batch rows per workgroup (2 WMMA m-subtiles)
#define THREADS  512            // 16 waves of 32
#define KSEC     256            // x staging section width (4 sections)
#define XSTR     (KSEC + 8)     // padded LDS row stride for x section (bf16)
#define HSTR     (HIDDEN + 8)   // padded LDS row stride for h (bf16)
#define NTSTRIDE ((size_t)16 * KTOT)  // elements between adjacent n-tiles' rows

typedef __bf16 bf16_t;
typedef bf16_t v16bf __attribute__((ext_vector_type(16)));
typedef bf16_t v8bf  __attribute__((ext_vector_type(8)));
typedef float  v8f   __attribute__((ext_vector_type(8)));

__device__ __forceinline__ bf16_t cvt_bf16(float f) {
  union { float f; uint32_t u; } v; v.f = f;
  uint32_t r = v.u + 0x7FFFu + ((v.u >> 16) & 1u);   // round-to-nearest-even
  union { uint16_t s; bf16_t b; } o; o.s = (uint16_t)(r >> 16);
  return o.b;
}

__device__ __forceinline__ float fast_sigmoid(float x) {
  float e = __builtin_amdgcn_exp2f(-x * 1.4426950408889634f);
  return __builtin_amdgcn_rcpf(1.0f + e);
}
__device__ __forceinline__ float fast_tanh(float x) {
  float e = __builtin_amdgcn_exp2f(-2.0f * x * 1.4426950408889634f);
  return (1.0f - e) * __builtin_amdgcn_rcpf(1.0f + e);
}

// Build a 16-element bf16 fragment from two 8-element (16-byte) runs.
__device__ __forceinline__ v16bf load_frag16(const bf16_t* p0, const bf16_t* p1) {
  v8bf a0 = *(const v8bf*)p0;
  v8bf a1 = *(const v8bf*)p1;
  v16bf r;
#pragma unroll
  for (int e = 0; e < 8; ++e) { r[e] = a0[e]; r[e + 8] = a1[e]; }
  return r;
}

// ---------------------------------------------------------------------------
// Kernel 1: convert + permute weights to bf16, combine biases.
// Row permutation: n -> gate = (n>>4)&3, j = ((n>>6)<<4)|(n&15), so each
// 16-wide N tile is a single gate and tiles cycle i,f,g,o per 16-j block
// (makes the LSTM cell update lane-local).
// wcat[d][n][k]: k<1024 from w_ih, k>=1024 from w_hh.
// ---------------------------------------------------------------------------
__global__ void convert_weights(const float* __restrict__ w_ih_l, const float* __restrict__ w_hh_l,
                                const float* __restrict__ b_ih_l, const float* __restrict__ b_hh_l,
                                const float* __restrict__ w_ih_r, const float* __restrict__ w_hh_r,
                                const float* __restrict__ b_ih_r, const float* __restrict__ b_hh_r,
                                bf16_t* __restrict__ wcat, float* __restrict__ biasc) {
  const int d = blockIdx.x >> 11;       // 2048 rows per direction
  const int n = blockIdx.x & (GATES - 1);
  const float* wih = d ? w_ih_r : w_ih_l;
  const float* whh = d ? w_hh_r : w_hh_l;
  const float* bih = d ? b_ih_r : b_ih_l;
  const float* bhh = d ? b_hh_r : b_hh_l;
  const int gate = (n >> 4) & 3;
  const int j    = ((n >> 6) << 4) | (n & 15);
  const int g    = gate * HIDDEN + j;
  bf16_t* dst = wcat + (size_t)(d * GATES + n) * KTOT;
  for (int k = threadIdx.x; k < KTOT; k += blockDim.x) {
    float v = (k < INPUT) ? wih[(size_t)g * INPUT + k]
                          : whh[(size_t)g * HIDDEN + (k - INPUT)];
    dst[k] = cvt_bf16(v);
  }
  if (threadIdx.x == 0) biasc[d * GATES + n] = bih[g] + bhh[g];
}

// Software-pipelined B-fragment sweep over 8 n-tiles: the load for tile i+1
// is issued before the WMMAs of tile i. Tile offsets are compile-time
// immediates off one per-lane base pointer (no per-tile address VALU).
__device__ __forceinline__ void mma_sweep(const bf16_t* bp, const v16bf& a0,
                                          const v16bf& a1, v8f acc0[8], v8f acc1[8]) {
  v16bf b_cur = load_frag16(bp, bp + 16);
#pragma unroll
  for (int i = 0; i < 8; ++i) {
    v16bf b_nxt = {};
    if (i < 7) {
      const bf16_t* bn = bp + (size_t)(i + 1) * NTSTRIDE;
      b_nxt = load_frag16(bn, bn + 16);
    }
    acc0[i] = __builtin_amdgcn_wmma_f32_16x16x32_bf16(
        false, a0, false, b_cur, (short)0, acc0[i], false, false);
    acc1[i] = __builtin_amdgcn_wmma_f32_16x16x32_bf16(
        false, a1, false, b_cur, (short)0, acc1[i], false, false);
    b_cur = b_nxt;
  }
}

// ---------------------------------------------------------------------------
// Kernel 2: fused bidirectional LSTM. One block = 32 batch rows, one
// direction, all 26 time steps. Per step: gates = [x_t | h] (32x1536, bf16)
// x Wcat^T (1536x2048, bf16) via v_wmma_f32_16x16x32_bf16 with two m-subtiles
// per wave (each B fragment feeds 2 WMMAs -> 2x L2 arithmetic intensity),
// then lane-local LSTM cell update (c stays in registers).
// ---------------------------------------------------------------------------
__global__ void __launch_bounds__(THREADS)
bilstm_kernel(const float*  __restrict__ x,      // [B][T][INPUT] f32
              const bf16_t* __restrict__ wcat,   // [2][GATES][KTOT] bf16
              const float*  __restrict__ biasc,  // [2][GATES]
              float*        __restrict__ out) {  // [B][T][2*HIDDEN] f32
  __shared__ bf16_t x_bf[MTILE * XSTR];          // one 256-wide K section
  __shared__ bf16_t h_bf[MTILE * HSTR];

  const int tid   = threadIdx.x;
  const int wave  = tid >> 5;
  const int lane  = tid & 31;
  const int l16   = lane & 15;
  const int khalf = lane >> 4;

  const int dir = blockIdx.x >> 6;                // 64 blocks per direction
  const int b0  = (blockIdx.x & 63) * MTILE;

  // h = 0 initial state
  for (int i = tid; i < MTILE * HSTR; i += THREADS) h_bf[i] = (bf16_t)0.0f;

  // Single per-lane weight base pointer; n-tile i lives at + i*NTSTRIDE.
  const bf16_t* bbase = wcat +
      ((size_t)dir * GATES + (size_t)wave * 128 + l16) * KTOT + khalf * 8;

  float biasr[8];
#pragma unroll
  for (int i = 0; i < 8; ++i)
    biasr[i] = biasc[dir * GATES + (wave * 8 + i) * 16 + l16];

  float creg[2][2][8];                            // [m-sub][half][v] cell state
#pragma unroll
  for (int s = 0; s < 2; ++s)
#pragma unroll
    for (int h = 0; h < 2; ++h)
#pragma unroll
      for (int v = 0; v < 8; ++v) creg[s][h][v] = 0.0f;

  for (int t = 0; t < T_FRAMES; ++t) {
    const int tt = dir ? (T_FRAMES - 1 - t) : t;  // reverse dir scans backwards

    v8f acc[2][8];                                // [m-sub][n-tile]
#pragma unroll
    for (int s = 0; s < 2; ++s)
#pragma unroll
      for (int i = 0; i < 8; ++i) acc[s][i] = (v8f){};

    // ---- x contribution: 4 staged sections of K=256 --------------------
    for (int ks = 0; ks < INPUT / KSEC; ++ks) {
      __syncthreads();                            // x_bf free / h_bf settled
      // Stage x section (32 x 256) f32 -> bf16 into LDS, coalesced.
      const float* xsrc = x + ((size_t)b0 * T_FRAMES + tt) * INPUT + ks * KSEC;
#pragma unroll
      for (int r = 0; r < (MTILE * KSEC) / THREADS; ++r) {
        int idx = tid + r * THREADS;
        int m = idx >> 8;                         // / KSEC
        int k = idx & (KSEC - 1);
        x_bf[m * XSTR + k] = cvt_bf16(xsrc[(size_t)m * T_FRAMES * INPUT + k]);
      }
      __syncthreads();

      const bf16_t* abase = x_bf + l16 * XSTR + khalf * 8;
      for (int kc = 0; kc < KSEC / 32; ++kc) {
        const bf16_t* ap0 = abase + kc * 32;
        const bf16_t* ap1 = ap0 + 16 * XSTR;
        v16bf a0 = load_frag16(ap0, ap0 + 16);
        v16bf a1 = load_frag16(ap1, ap1 + 16);
        mma_sweep(bbase + (ks * (KSEC / 32) + kc) * 32, a0, a1, acc[0], acc[1]);
      }
    }

    // ---- h contribution: K = 512 from LDS ------------------------------
    {
      const bf16_t* abase = h_bf + l16 * HSTR + khalf * 8;
      for (int kc = 0; kc < HIDDEN / 32; ++kc) {
        const bf16_t* ap0 = abase + kc * 32;
        const bf16_t* ap1 = ap0 + 16 * HSTR;
        v16bf a0 = load_frag16(ap0, ap0 + 16);
        v16bf a1 = load_frag16(ap1, ap1 + 16);
        mma_sweep(bbase + (INPUT / 32 + kc) * 32, a0, a1, acc[0], acc[1]);
      }
    }
    __syncthreads();                              // all waves done reading h_bf

    // ---- LSTM cell: tiles i = half*4 + {i,f,g,o} share (m, j) per lane --
#pragma unroll
    for (int s = 0; s < 2; ++s) {
#pragma unroll
      for (int half = 0; half < 2; ++half) {
        const int j = (wave * 2 + half) * 16 + l16;   // hidden index 0..511
#pragma unroll
        for (int v = 0; v < 8; ++v) {
          const int m = s * 16 + v + 8 * khalf;       // batch row within tile
          float gi = acc[s][half * 4 + 0][v] + biasr[half * 4 + 0];
          float gf = acc[s][half * 4 + 1][v] + biasr[half * 4 + 1];
          float gg = acc[s][half * 4 + 2][v] + biasr[half * 4 + 2];
          float go = acc[s][half * 4 + 3][v] + biasr[half * 4 + 3];
          float c = fast_sigmoid(gf) * creg[s][half][v] +
                    fast_sigmoid(gi) * fast_tanh(gg);
          float h = fast_sigmoid(go) * fast_tanh(c);
          creg[s][half][v] = c;
          h_bf[m * HSTR + j] = cvt_bf16(h);           // next step's A matrix
          out[(((size_t)(b0 + m)) * T_FRAMES + tt) * (2 * HIDDEN) + dir * HIDDEN + j] = h;
        }
      }
    }
  }
}

extern "C" void kernel_launch(void* const* d_in, const int* in_sizes, int n_in,
                              void* d_out, int out_size, void* d_ws, size_t ws_size,
                              hipStream_t stream) {
  const float* x      = (const float*)d_in[0];
  const float* w_ih_l = (const float*)d_in[1];
  const float* w_hh_l = (const float*)d_in[2];
  const float* b_ih_l = (const float*)d_in[3];
  const float* b_hh_l = (const float*)d_in[4];
  const float* w_ih_r = (const float*)d_in[5];
  const float* w_hh_r = (const float*)d_in[6];
  const float* b_ih_r = (const float*)d_in[7];
  const float* b_hh_r = (const float*)d_in[8];

  // Workspace: bf16 concatenated+permuted weights, then combined biases.
  bf16_t* wcat  = (bf16_t*)d_ws;                       // 2*2048*1536*2 B = 12.6 MB
  float*  biasc = (float*)((char*)d_ws +
                  (size_t)2 * GATES * KTOT * sizeof(bf16_t)); // +16 KB

  convert_weights<<<2 * GATES, 256, 0, stream>>>(
      w_ih_l, w_hh_l, b_ih_l, b_hh_l, w_ih_r, w_hh_r, b_ih_r, b_hh_r,
      wcat, biasc);

  bilstm_kernel<<<2 * (BATCH / MTILE), THREADS, 0, stream>>>(
      x, wcat, biasc, (float*)d_out);
}
